// ContextAwareModel_11441792877247
// MI455X (gfx1250) — compile-verified, matrix operand order checked by reading, not run on hardware
//
#include <hip/hip_runtime.h>
#include <hip/hip_bf16.h>
#include <stdint.h>

// ---------------------------------------------------------------------------
// Problem constants (from the reference)
// ---------------------------------------------------------------------------
#define SEQ 512
#define BAT 64
#define EMB 1024
#define HID 1024
#define G4  4096   // 4*H gate width

typedef __bf16 bf16_t;
typedef __attribute__((ext_vector_type(16))) __bf16 bf16x16;
typedef __attribute__((ext_vector_type(4)))  __bf16 bf16x4;
typedef __attribute__((ext_vector_type(8)))  float  f32x8;

#define DEV static __device__ __forceinline__

union FragU { uint4 q[2]; bf16x16 v; };

// Load one 16x32 bf16 fragment (A-layout per CDNA5 ISA 7.12.2):
// lanes 0-15 hold K=[0..7] in v0..3 and K=[16..23] in v4..7 (row M = lane);
// lanes 16-31 hold K=[8..15] and K=[24..31] (row M = lane-16).
// Both halves are contiguous 16-byte chunks -> two global_load_b128.
// W is stored row-major [N,K], so a B-matrix column n == W row n and uses
// the exact same load pattern.
DEV bf16x16 load_frag_k32(const bf16_t* __restrict__ base, int ld, int lane) {
  const int r  = lane & 15;
  const int hs = lane >> 4;
  const bf16_t* p = base + (size_t)r * ld + hs * 8;
  FragU f;
  f.q[0] = *reinterpret_cast<const uint4*>(p);
  f.q[1] = *reinterpret_cast<const uint4*>(p + 16);
  return f.v;
}

DEV f32x8 wmma_bf16(bf16x16 a, bf16x16 b, f32x8 c) {
  // (neg_a, A, neg_b, B, c_mod, C, reuse_a, reuse_b)
  return __builtin_amdgcn_wmma_f32_16x16x32_bf16(false, a, false, b,
                                                 (short)0, c, false, false);
}

// ---------------------------------------------------------------------------
// fp32 -> bf16 bulk conversion (n is a multiple of 4)
// ---------------------------------------------------------------------------
__global__ void f32_to_bf16_kernel(const float* __restrict__ in,
                                   bf16_t* __restrict__ out, size_t n) {
  size_t i = ((size_t)blockIdx.x * blockDim.x + threadIdx.x) * 4;
  if (i + 4 <= n) {
    const float4 f = *reinterpret_cast<const float4*>(in + i);
    bf16x4 o = { (bf16_t)f.x, (bf16_t)f.y, (bf16_t)f.z, (bf16_t)f.w };
    *reinterpret_cast<bf16x4*>(out + i) = o;
  }
}

// ---------------------------------------------------------------------------
// Embedding gather + convert: x[s][b][:] = bf16(emb[ids[b][s]][:])
// ---------------------------------------------------------------------------
__global__ __launch_bounds__(256)
void embed_gather_kernel(const int* __restrict__ ids,
                         const float* __restrict__ emb,
                         bf16_t* __restrict__ x) {
  const int s = blockIdx.x, b = blockIdx.y;
  const int id = ids[b * SEQ + s];
  const float* src = emb + (size_t)id * EMB;
  bf16_t* dst = x + ((size_t)s * BAT + b) * EMB;
  const int e = threadIdx.x * 4;                      // 256 thr * 4 = 1024
  const float4 f = *reinterpret_cast<const float4*>(src + e);
  bf16x4 o = { (bf16_t)f.x, (bf16_t)f.y, (bf16_t)f.z, (bf16_t)f.w };
  *reinterpret_cast<bf16x4*>(dst + e) = o;
}

// ---------------------------------------------------------------------------
// Batched input projection for both directions:
//   C[z][m][n] = sum_k A[m][k] * W[z][n][k] + bias[z][n]     (bf16 out, f32 acc)
// block = 256 thr (8 waves), tile 64(M) x 256(N); wave tile 32x64 = 2x4 WMMA.
// ---------------------------------------------------------------------------
__global__ __launch_bounds__(256)
void gemm_proj_kernel(const bf16_t* __restrict__ A, const bf16_t* __restrict__ W,
                      const float* __restrict__ bias, bf16_t* __restrict__ C,
                      int M, int N, int K) {
  const int z = blockIdx.z;
  W    += (size_t)z * N * K;
  bias += (size_t)z * N;
  C    += (size_t)z * M * N;

  const int lane = threadIdx.x & 31, w = threadIdx.x >> 5;
  const int m0 = blockIdx.y * 64  + (w & 1) * 32;
  const int n0 = blockIdx.x * 256 + (w >> 1) * 64;
  const int r = lane & 15, hs = lane >> 4;

  const f32x8 zero = {0.f,0.f,0.f,0.f,0.f,0.f,0.f,0.f};
  f32x8 acc[2][4];
  #pragma unroll
  for (int i = 0; i < 2; ++i)
    #pragma unroll
    for (int j = 0; j < 4; ++j) acc[i][j] = zero;

  for (int kb = 0; kb < K; kb += 32) {
    // hint L2 prefetch of next A chunk (global_prefetch_b8)
    if (kb + 32 < K)
      __builtin_prefetch(A + (size_t)m0 * K + kb + 32, 0, 1);
    bf16x16 a0 = load_frag_k32(A + (size_t)m0 * K + kb,        K, lane);
    bf16x16 a1 = load_frag_k32(A + (size_t)(m0 + 16) * K + kb, K, lane);
    #pragma unroll
    for (int j = 0; j < 4; ++j) {
      bf16x16 b = load_frag_k32(W + (size_t)(n0 + j * 16) * K + kb, K, lane);
      acc[0][j] = wmma_bf16(a0, b, acc[0][j]);
      acc[1][j] = wmma_bf16(a1, b, acc[1][j]);
    }
  }

  // Epilogue: C/D layout -> lane holds column n = n0+j*16+r, rows hs*8+v.
  #pragma unroll
  for (int j = 0; j < 4; ++j) {
    const float bv = bias[n0 + j * 16 + r];
    #pragma unroll
    for (int i = 0; i < 2; ++i) {
      #pragma unroll
      for (int v = 0; v < 8; ++v) {
        const int m = m0 + i * 16 + hs * 8 + v;
        C[(size_t)m * N + n0 + j * 16 + r] = (bf16_t)(acc[i][j][v] + bv);
      }
    }
  }
}

// ---------------------------------------------------------------------------
// Persistent BiLSTM scan (one layer, BOTH directions concurrently).
// 128 blocks: block = (dir, bk); block permanently owns h-columns
// [bk*16, bk*16+16). The block's W_hh slice (64 gate rows x HID = 128 KB bf16)
// is staged ONCE into LDS via CDNA5 async copy (global_load_async_to_lds_b128,
// ASYNCcnt), so the 512-step dependency chain runs at LDS bandwidth, not L2.
// Per timestep: 64(B) x 64 gate slice = h_prev @ W_hh^T via WMMA (A from the
// L2-resident ping-pong h buffer, B from LDS), add precomputed xw gates,
// LSTM cell update with c-state persistent in LDS, then one device-wide
// barrier (atomic counter + s_sleep spin). One barrier/step suffices because
// h is double-buffered.
// ---------------------------------------------------------------------------
#define SCAN_BLOCKS_PER_DIR 64
#define SCAN_BLOCKS (2 * SCAN_BLOCKS_PER_DIR)
#define LDW (HID + 8)                 // padded LDS row stride (elements):
                                      // 2064 B -> lane r hits banks 4r..4r+3
#define WLDS_BYTES (64 * LDW * (int)sizeof(bf16_t))   // 132096 B dynamic LDS

__global__ __launch_bounds__(256)
void lstm_scan_kernel(const bf16_t* __restrict__ xw,  // [2][S][B][4H] bias folded
                      const bf16_t* __restrict__ whh, // [2][4H][H]
                      bf16_t* __restrict__ hbuf,      // [2][2][B][H] pre-zeroed
                      bf16_t* __restrict__ y,         // [S][B][2H]
                      unsigned* __restrict__ bar)     // pre-zeroed counter
{
  extern __shared__ __align__(16) bf16_t wlds[];      // [64][LDW] W_hh slice

  const int dir = blockIdx.x / SCAN_BLOCKS_PER_DIR;
  const int bk  = blockIdx.x % SCAN_BLOCKS_PER_DIR;
  const int tid = threadIdx.x, lane = tid & 31, w = tid >> 5;
  const int r = lane & 15, hs = lane >> 4;
  const int mt = w & 3;           // which 16-row batch tile
  const int q0 = (w >> 2) * 2;    // quadrant pair: waves 0-3 -> {i,f}, 4-7 -> {g,o}
  const int hc0 = bk * 16;
  const unsigned nblk = gridDim.x;

  const bf16_t* Wd  = whh + (size_t)dir * G4 * HID;
  const bf16_t* xwD = xw  + (size_t)dir * SEQ * BAT * G4;
  bf16_t* h0 = hbuf + (size_t)dir * 2 * BAT * HID;   // parity 0
  bf16_t* h1 = h0 + (size_t)BAT * HID;               // parity 1

  __shared__ float gs[BAT][68];    // [b][q*16 + c] gate slice (padded)
  __shared__ float cs[BAT][16];    // persistent cell state for owned columns
  for (int e = tid; e < BAT * 16; e += 256) cs[e >> 4][e & 15] = 0.0f;

  // ---- one-time async staging of this block's W_hh slice into LDS ----
  // Local row lr in [0,64): gate quadrant q = lr>>4, owned col c = lr&15,
  // global W row = q*HID + hc0 + c. 128 x b128 transfers per row.
  for (int mq = tid; mq < 64 * (HID / 8); mq += 256) {
    const int lr   = mq >> 7;               // HID/8 = 128 uint4 per row
    const int off  = mq & 127;
    const int grow = (lr >> 4) * HID + hc0 + (lr & 15);
    const bf16_t* gp = Wd + (size_t)grow * HID + (size_t)off * 8;
    const unsigned laddr =
        (unsigned)(size_t)(const void*)(wlds + lr * LDW + off * 8);
    asm volatile("global_load_async_to_lds_b128 %0, %1, off"
                 :: "v"(laddr), "v"(gp) : "memory");
  }
  asm volatile("s_wait_asynccnt 0x0" ::: "memory");
  __syncthreads();

  const f32x8 zero = {0.f,0.f,0.f,0.f,0.f,0.f,0.f,0.f};
  // LDS base pointers for this wave's two B tiles (quadrants q0, q0+1):
  // B-fragment column n == local row (q*16 + r), contiguous in K.
  const bf16_t* Bq0 = wlds + (size_t)(( q0      * 16) + r) * LDW + hs * 8;
  const bf16_t* Bq1 = wlds + (size_t)(((q0 + 1) * 16) + r) * LDW + hs * 8;

  for (int t = 0; t < SEQ; ++t) {
    const bf16_t* hin  = (t & 1) ? h1 : h0;
    bf16_t*       hout = (t & 1) ? h0 : h1;

    f32x8 c0 = zero, c1 = zero;
    const bf16_t* Ab = hin + (size_t)(mt * 16) * HID;
    for (int kb = 0; kb < HID; kb += 32) {
      bf16x16 a = load_frag_k32(Ab + kb, HID, lane);
      FragU b0, b1;
      b0.q[0] = *reinterpret_cast<const uint4*>(Bq0 + kb);
      b0.q[1] = *reinterpret_cast<const uint4*>(Bq0 + kb + 16);
      b1.q[0] = *reinterpret_cast<const uint4*>(Bq1 + kb);
      b1.q[1] = *reinterpret_cast<const uint4*>(Bq1 + kb + 16);
      c0 = wmma_bf16(a, b0.v, c0);
      c1 = wmma_bf16(a, b1.v, c1);
    }

    // add precomputed input-projection gates and stage into LDS
    const bf16_t* xwt = xwD + (size_t)t * BAT * G4;
    #pragma unroll
    for (int v = 0; v < 8; ++v) {
      const int b = mt * 16 + hs * 8 + v;
      const size_t rowoff = (size_t)b * G4 + hc0 + r;
      gs[b][ q0      * 16 + r] = c0[v] + (float)xwt[rowoff + (size_t) q0      * HID];
      gs[b][(q0 + 1) * 16 + r] = c1[v] + (float)xwt[rowoff + (size_t)(q0 + 1) * HID];
    }
    __syncthreads();

    // LSTM cell update for owned columns (torch gate order i,f,g,o)
    for (int e = tid; e < BAT * 16; e += 256) {
      const int b = e >> 4, c = e & 15;
      const float ig = gs[b][c],      fg = gs[b][16 + c];
      const float gg = gs[b][32 + c], og = gs[b][48 + c];
      const float i_ = 1.0f / (1.0f + __expf(-ig));
      const float f_ = 1.0f / (1.0f + __expf(-fg));
      const float o_ = 1.0f / (1.0f + __expf(-og));
      const float cc = f_ * cs[b][c] + i_ * tanhf(gg);
      cs[b][c] = cc;
      const bf16_t hh = (bf16_t)(o_ * tanhf(cc));
      hout[(size_t)b * HID + hc0 + c] = hh;
      y[((size_t)t * BAT + b) * (2 * HID) + (size_t)dir * HID + hc0 + c] = hh;
    }

    // device-wide barrier: release h writes, arrive, spin, acquire
    __threadfence();
    __syncthreads();
    if (tid == 0) {
      atomicAdd(bar, 1u);
      const unsigned want = (unsigned)(t + 1) * nblk;
      while (atomicAdd(bar, 0u) < want) __builtin_amdgcn_s_sleep(1);
    }
    __syncthreads();
    __threadfence();
  }
}

// ---------------------------------------------------------------------------
// Classifier: out[b] = sigmoid( dot(y[tgt[b]][b][:], w_cls) + b_cls )
// ---------------------------------------------------------------------------
__global__ __launch_bounds__(256)
void classify_kernel(const bf16_t* __restrict__ y,   // [S][B][2H]
                     const int* __restrict__ tgt,    // [B]
                     const float* __restrict__ wc,   // [2H]
                     const float* __restrict__ bc,   // [1]
                     float* __restrict__ out) {
  const int b = blockIdx.x;
  const int t = tgt[b];
  const bf16_t* v = y + ((size_t)t * BAT + b) * (2 * HID);
  float acc = 0.0f;
  for (int i = threadIdx.x; i < 2 * HID; i += 256)
    acc += (float)v[i] * wc[i];
  #pragma unroll
  for (int off = 16; off > 0; off >>= 1) acc += __shfl_down(acc, off, 32);
  __shared__ float red[8];
  if ((threadIdx.x & 31) == 0) red[threadIdx.x >> 5] = acc;
  __syncthreads();
  if (threadIdx.x == 0) {
    float s = 0.0f;
    #pragma unroll
    for (int i = 0; i < 8; ++i) s += red[i];
    s += bc[0];
    out[b] = 1.0f / (1.0f + __expf(-s));
  }
}

// ---------------------------------------------------------------------------
// Host-side orchestration (graph-capture safe: kernels + hipMemsetAsync only)
// ---------------------------------------------------------------------------
extern "C" void kernel_launch(void* const* d_in, const int* in_sizes, int n_in,
                              void* d_out, int out_size, void* d_ws, size_t ws_size,
                              hipStream_t stream) {
  const int*   ids   = (const int*)  d_in[0];
  const int*   tgt   = (const int*)  d_in[1];
  const float* emb   = (const float*)d_in[2];
  const float* wih0f = (const float*)d_in[3];
  const float* whh0f = (const float*)d_in[4];
  const float* b0    = (const float*)d_in[5];
  const float* wih1f = (const float*)d_in[6];
  const float* whh1f = (const float*)d_in[7];
  const float* b1    = (const float*)d_in[8];
  const float* wcls  = (const float*)d_in[9];
  const float* bcls  = (const float*)d_in[10];
  float* out = (float*)d_out;

  // allow >64KB dynamic LDS for the scan kernel (not a stream op; capture-safe)
  hipFuncSetAttribute((const void*)lstm_scan_kernel,
                      hipFuncAttributeMaxDynamicSharedMemorySize, WLDS_BYTES);

  // workspace bump allocator (256B aligned)
  char* p = (char*)d_ws;
  auto take = [&](size_t bytes) -> char* {
    char* r = p; p += (bytes + 255) & ~(size_t)255; return r;
  };
  const size_t nWih0 = (size_t)2 * G4 * EMB;
  const size_t nWhh  = (size_t)2 * G4 * HID;
  const size_t nWih1 = (size_t)2 * G4 * 2 * HID;
  bf16_t* wih0 = (bf16_t*)take(nWih0 * sizeof(bf16_t));
  bf16_t* whh0 = (bf16_t*)take(nWhh  * sizeof(bf16_t));
  bf16_t* wih1 = (bf16_t*)take(nWih1 * sizeof(bf16_t));
  bf16_t* whh1 = (bf16_t*)take(nWhh  * sizeof(bf16_t));
  bf16_t* xbf  = (bf16_t*)take((size_t)SEQ * BAT * EMB * sizeof(bf16_t));
  bf16_t* xwbf = (bf16_t*)take((size_t)2 * SEQ * BAT * G4 * sizeof(bf16_t));
  bf16_t* y0   = (bf16_t*)take((size_t)SEQ * BAT * 2 * HID * sizeof(bf16_t));
  bf16_t* y1   = (bf16_t*)take((size_t)SEQ * BAT * 2 * HID * sizeof(bf16_t));
  bf16_t* hbuf = (bf16_t*)take((size_t)2 * 2 * BAT * HID * sizeof(bf16_t));
  unsigned* bar = (unsigned*)take(256);

  auto cvt = [&](const float* src, bf16_t* dst, size_t n) {
    const unsigned blocks = (unsigned)((n / 4 + 255) / 256);
    f32_to_bf16_kernel<<<blocks, 256, 0, stream>>>(src, dst, n);
  };
  cvt(wih0f, wih0, nWih0);
  cvt(whh0f, whh0, nWhh);
  cvt(wih1f, wih1, nWih1);
  cvt(whh1f, whh1, nWhh);

  embed_gather_kernel<<<dim3(SEQ, BAT), 256, 0, stream>>>(ids, emb, xbf);

  hipMemsetAsync(bar, 0, 256, stream);
  hipMemsetAsync(hbuf, 0, (size_t)2 * 2 * BAT * HID * sizeof(bf16_t), stream);

  // Layer 0: time-parallel input projection, then persistent scan
  gemm_proj_kernel<<<dim3(G4 / 256, (SEQ * BAT) / 64, 2), 256, 0, stream>>>(
      xbf, wih0, b0, xwbf, SEQ * BAT, G4, EMB);
  lstm_scan_kernel<<<SCAN_BLOCKS, 256, WLDS_BYTES, stream>>>(
      xwbf, whh0, hbuf, y0, bar + 0);

  // Layer 1 (input = y0, K = 2H); xw buffer reused
  hipMemsetAsync(hbuf, 0, (size_t)2 * 2 * BAT * HID * sizeof(bf16_t), stream);
  gemm_proj_kernel<<<dim3(G4 / 256, (SEQ * BAT) / 64, 2), 256, 0, stream>>>(
      y0, wih1, b1, xwbf, SEQ * BAT, G4, 2 * HID);
  lstm_scan_kernel<<<SCAN_BLOCKS, 256, WLDS_BYTES, stream>>>(
      xwbf, whh1, hbuf, y1, bar + 16);

  classify_kernel<<<BAT, 256, 0, stream>>>(y1, tgt, wcls, bcls, out);

  (void)in_sizes; (void)n_in; (void)out_size; (void)ws_size;
}
